// MLP_J_47571057771078
// MI455X (gfx1250) — compile-verified, matrix operand order checked by reading, not run on hardware
//
#include <hip/hip_runtime.h>
#include <hip/hip_bf16.h>

typedef __attribute__((ext_vector_type(16))) _Float16 v16h;
typedef __attribute__((ext_vector_type(8)))  float    v8f;

#define D_DIM 256
#define H_DIM 1024
#define B_DIM 65536
#define WAVES_PER_WG 8
#define ROWS_PER_WAVE 32           // two 16-row WMMA blocks per wave
#define TILE_HALFS (8 * 32 * 16)   // one staged B tile: 8 K-tiles x 32 lanes x 16 halfs

// Kernel 1: rowsum[b] = sum_n silu( (x @ A)[b, n] ) via f16 WMMA, f32 accum.
// The [B, h] intermediate never touches memory. Each wave owns 32 rows (two
// A-fragment sets) so every staged weight tile / LDS fragment load is
// amortized over 2 WMMA chains, halving L2 and LDS traffic per FLOP.
__global__ __launch_bounds__(256) void gemm_silu_rowsum(
    const float* __restrict__ x, const float* __restrict__ w,
    float* __restrict__ rowsum) {
  // Double-buffered B-operand staging, pre-swizzled into WMMA fragment layout.
  __shared__ __align__(32) _Float16 lds_b[2][TILE_HALFS];

  const int tid    = threadIdx.x;
  const int wave   = tid >> 5;
  const int lane   = tid & 31;
  const int half16 = lane >> 4;
  const int lane16 = lane & 15;

  const int r0 = (blockIdx.x * WAVES_PER_WG + wave) * ROWS_PER_WAVE;

  // ---- Register-resident x fragments (A-operand), ISA 16-bit A layout:
  // lane (half,m): elems 0..7 hold K = kbase+0..7, elems 8..15 hold
  // K = kbase+16..23, kbase = 32*kt + 8*half.
  v16h afragA[8], afragB[8];
#pragma unroll
  for (int blk = 0; blk < 2; ++blk) {
    v16h* afrag = blk ? afragB : afragA;
    const float* xrow = x + (size_t)(r0 + blk * 16 + lane16) * D_DIM;
#pragma unroll
    for (int kt = 0; kt < 8; ++kt) {
      const int kbase = kt * 32 + half16 * 8;
      const float4* p0 = (const float4*)(xrow + kbase);
      const float4* p1 = (const float4*)(xrow + kbase + 16);
      float4 a0 = p0[0], a1 = p0[1];
      float4 b0 = p1[0], b1 = p1[1];
      afrag[kt][0]  = (_Float16)a0.x; afrag[kt][1]  = (_Float16)a0.y;
      afrag[kt][2]  = (_Float16)a0.z; afrag[kt][3]  = (_Float16)a0.w;
      afrag[kt][4]  = (_Float16)a1.x; afrag[kt][5]  = (_Float16)a1.y;
      afrag[kt][6]  = (_Float16)a1.z; afrag[kt][7]  = (_Float16)a1.w;
      afrag[kt][8]  = (_Float16)b0.x; afrag[kt][9]  = (_Float16)b0.y;
      afrag[kt][10] = (_Float16)b0.z; afrag[kt][11] = (_Float16)b0.w;
      afrag[kt][12] = (_Float16)b1.x; afrag[kt][13] = (_Float16)b1.y;
      afrag[kt][14] = (_Float16)b1.z; afrag[kt][15] = (_Float16)b1.w;
    }
  }

  // Stage-time swizzle indices (p-invariant): this thread owns k rows
  // {p*32 + krow, p*32 + krow + 1} of column n for every K-tile p.
  const int krow = (tid >> 4) * 2;   // even kk in [0, 32)
  const int n    = tid & 15;
  const int g     = krow >> 3;
  const int bhalf = g & 1;
  const int bv    = ((g >> 1) << 2) + ((krow & 7) >> 1);
  const int blane = bhalf * 16 + n;
  const int lds_base = blane * 16 + bv * 2;         // + p*512 per K-tile
  const size_t g_base = (size_t)krow * H_DIM + n;   // + nt*16 + p*32*H_DIM

  // Stage weight tile nt into lds_b[buf]: one packed-f16 ds_store_b32 per
  // K-tile per thread.
  auto stage = [&](int nt, int buf) {
    const float* wp = w + g_base + nt * 16;
#pragma unroll
    for (int p = 0; p < 8; ++p) {
      float v0 = wp[(size_t)p * 32 * H_DIM];
      float v1 = wp[(size_t)p * 32 * H_DIM + H_DIM];
      auto pk = __builtin_amdgcn_cvt_pkrtz(v0, v1);  // v_cvt_pk_rtz_f16_f32
      *(unsigned int*)&lds_b[buf][p * 512 + lds_base] =
          __builtin_bit_cast(unsigned int, pk);
    }
  };

  float rsumA[8], rsumB[8];
#pragma unroll
  for (int v = 0; v < 8; ++v) { rsumA[v] = 0.0f; rsumB[v] = 0.0f; }

  stage(0, 0);
  __syncthreads();

  for (int nt = 0; nt < 64; ++nt) {
    const int buf = nt & 1;
    if (nt + 1 < 64) stage(nt + 1, buf ^ 1);  // overlaps with WMMA below

    // Two independent accumulator chains share each B fragment load.
    v8f cA = {}, cB = {};
#pragma unroll
    for (int kt = 0; kt < 8; ++kt) {
      v16h bfrag = *(const v16h*)&lds_b[buf][(kt * 32 + lane) * 16];
      cA = __builtin_amdgcn_wmma_f32_16x16x32_f16(
          false, afragA[kt], false, bfrag, (short)0, cA, false, false);
      cB = __builtin_amdgcn_wmma_f32_16x16x32_f16(
          false, afragB[kt], false, bfrag, (short)0, cB, false, false);
    }

    // silu + accumulate: sigmoid via hardware v_rcp_f32 (no IEEE div chain).
#pragma unroll
    for (int v = 0; v < 8; ++v) {
      const float zA = cA[v];
      rsumA[v] += zA * __builtin_amdgcn_rcpf(1.0f + __expf(-zA));
      const float zB = cB[v];
      rsumB[v] += zB * __builtin_amdgcn_rcpf(1.0f + __expf(-zB));
    }
    __syncthreads();  // staging of buf^1 done; compute of buf done
  }

  // ---- Reduce over the 16 columns in this lane group (C layout:
  // VGPR v -> row M = v + 8*half; masks 1,2,4,8 stay inside the group).
#pragma unroll
  for (int v = 0; v < 8; ++v) {
    float sA = rsumA[v];
    sA += __shfl_xor(sA, 1);
    sA += __shfl_xor(sA, 2);
    sA += __shfl_xor(sA, 4);
    sA += __shfl_xor(sA, 8);
    float sB = rsumB[v];
    sB += __shfl_xor(sB, 1);
    sB += __shfl_xor(sB, 2);
    sB += __shfl_xor(sB, 4);
    sB += __shfl_xor(sB, 8);
    if (lane16 == 0) {
      rowsum[r0 + half16 * 8 + v]      = sA;
      rowsum[r0 + 16 + half16 * 8 + v] = sB;
    }
  }
}

// Kernel 2: bandwidth-bound fill of out[1024][256][256]; one nonzero per row:
//   i < 128 : out[g, i, 128+i] = +rowsum[g*64 + i/2]
//   i >= 128: out[g, i, i-128] = -rowsum[g*64 + (i-128)/2]
__global__ __launch_bounds__(256) void fill_output(
    const float* __restrict__ rowsum, float* __restrict__ out) {
  const int t = blockIdx.x * blockDim.x + threadIdx.x;  // 0 .. 4M-1
  const int part  = t & 15;
  const int rowid = t >> 4;        // g*256 + i
  const int i = rowid & 255;
  const int g = rowid >> 8;
  const int c0 = part * 16;

  float vals[16];
#pragma unroll
  for (int j = 0; j < 16; ++j) vals[j] = 0.0f;

  int c;
  float v;
  if (i < 128) {
    c = 128 + i;
    v = rowsum[g * 64 + (i >> 1)];
  } else {
    c = i - 128;
    v = -rowsum[g * 64 + ((i - 128) >> 1)];
  }
  if (c >= c0 && c < c0 + 16) vals[c - c0] = v;

  float4* dst = (float4*)(out + (size_t)rowid * 256 + c0);
  dst[0] = make_float4(vals[0], vals[1], vals[2], vals[3]);
  dst[1] = make_float4(vals[4], vals[5], vals[6], vals[7]);
  dst[2] = make_float4(vals[8], vals[9], vals[10], vals[11]);
  dst[3] = make_float4(vals[12], vals[13], vals[14], vals[15]);
}

extern "C" void kernel_launch(void* const* d_in, const int* in_sizes, int n_in,
                              void* d_out, int out_size, void* d_ws, size_t ws_size,
                              hipStream_t stream) {
  const float* x = (const float*)d_in[0];   // [65536, 256]
  const float* A = (const float*)d_in[1];   // [256, 1024]
  float* out     = (float*)d_out;           // [1024, 256, 256]
  float* rowsum  = (float*)d_ws;            // [65536] floats (256 KB scratch)

  // 65536 rows / (32 rows/wave * 8 waves/block) = 256 blocks
  gemm_silu_rowsum<<<256, 256, 0, stream>>>(x, A, rowsum);

  // 1024*256 rows * 16 chunks/row = 4,194,304 threads
  fill_output<<<16384, 256, 0, stream>>>(rowsum, out);
}
